// MultiHeadAttentionAwareTemporalContext_q1d_k1d_82566451299284
// MI455X (gfx1250) — matrix operand order repaired and checked by reading, not compiled, not run on hardware
//
#include <hip/hip_runtime.h>

#define B_  8
#define N_  300
#define T_  48
#define DM_ 512
#define H_  8
#define HD_ 64

typedef __attribute__((ext_vector_type(16))) __bf16 v16bf;
typedef __attribute__((ext_vector_type(8)))  __bf16 v8bf;
typedef __attribute__((ext_vector_type(8)))  float  v8f;

__device__ __forceinline__ v8f zero8() {
    v8f z;
#pragma unroll
    for (int e = 0; e < 8; ++e) z[e] = 0.0f;
    return z;
}

// A-fragment (16x32 bf16): lane<16 holds K kb..kb+15,
// lane>=16 holds K kb+16..kb+31.  aoff = (lane<16)?0:16.
__device__ __forceinline__ v16bf load_afrag(const __bf16* row, int kb, int aoff) {
    const __bf16* p = row + kb + aoff;
    v8bf lo = *(const v8bf*)(p);
    v8bf hi = *(const v8bf*)(p + 8);
    v16bf r;
#pragma unroll
    for (int e = 0; e < 8; ++e) { r[e] = lo[e]; r[8 + e] = hi[e]; }
    return r;
}

// B-fragment (32x16 bf16): lane<16 holds K kb..kb+15, lane>=16 holds K kb+16..kb+31.
// boff = (lane<16)?0:16.
__device__ __forceinline__ v16bf load_bfrag(const __bf16* row, int kb, int boff) {
    const __bf16* p = row + kb + boff;
    v8bf lo = *(const v8bf*)(p);
    v8bf hi = *(const v8bf*)(p + 8);
    v16bf r;
#pragma unroll
    for (int e = 0; e < 8; ++e) { r[e] = lo[e]; r[8 + e] = hi[e]; }
    return r;
}

__device__ __forceinline__ int segid(int t) { return (t < 12) ? 0 : ((t < 24) ? 1 : 2); }

// ---------------------------------------------------------------------------
// Kernel 0: repack weights f32 -> bf16.
// conv weights (o,i,1,3) -> [tap][o][i]; fc/proj (o,i) -> [o][i].
// ---------------------------------------------------------------------------
__global__ __launch_bounds__(256) void repack_weights(
    const float* __restrict__ cqw, const float* __restrict__ ckw,
    const float* __restrict__ fcw, const float* __restrict__ pjw,
    __bf16* __restrict__ wq, __bf16* __restrict__ wk,
    __bf16* __restrict__ wfc, __bf16* __restrict__ wpj)
{
    int idx = blockIdx.x * 256 + threadIdx.x;
    int which = blockIdx.y;
    if (which < 2) {
        // idx = tap*512*512 + o*512 + i  (786432 total)
        const float* src = which ? ckw : cqw;
        __bf16* dst = which ? wk : wq;
        int i   = idx & 511;
        int o   = (idx >> 9) & 511;
        int tap = idx >> 18;
        dst[idx] = (__bf16)src[((size_t)o * DM_ + i) * 3 + tap];
    } else {
        if (idx >= DM_ * DM_) return;
        const float* src = (which == 2) ? fcw : pjw;
        __bf16* dst = (which == 2) ? wfc : wpj;
        dst[idx] = (__bf16)src[idx];
    }
}

// ---------------------------------------------------------------------------
// Kernel 1: Q/K temporal-conv projection + V fc projection -> bf16 (b,n,t,o).
// One workgroup per (b*n, which).  which: 0=q(conv) 1=k(conv) 2=v(fc).
// The 1x3 conv is 3 shifted GEMMs; segment-boundary padding is handled by
// redirecting invalid source rows to a dedicated all-zero LDS row (row 48),
// so the inner loop has NO exec-mask branching.
// ---------------------------------------------------------------------------
__global__ __launch_bounds__(256) void proj_qkv(
    const float* __restrict__ query, const float* __restrict__ key_in,
    const float* __restrict__ value,
    const float* __restrict__ qbias, const float* __restrict__ kbias,
    const float* __restrict__ fcbias,
    const __bf16* __restrict__ wq, const __bf16* __restrict__ wk,
    const __bf16* __restrict__ wfc,
    __bf16* __restrict__ qo, __bf16* __restrict__ ko, __bf16* __restrict__ vo)
{
    __shared__ __attribute__((aligned(16))) __bf16 xs[(T_ + 1) * DM_];   // 49 rows, row 48 = zeros

    const int which = blockIdx.x % 3;
    const int bn    = blockIdx.x / 3;

    const float*  src;  const float* bias; const __bf16* w; __bf16* dst; int ntaps;
    if (which == 0)      { src = query;  bias = qbias;  w = wq;  dst = qo; ntaps = 3; }
    else if (which == 1) { src = key_in; bias = kbias;  w = wk;  dst = ko; ntaps = 3; }
    else                 { src = value;  bias = fcbias; w = wfc; dst = vo; ntaps = 1; }

    // stage the 48x512 input tile to LDS as bf16, plus a zero row
    const float* sp = src + (size_t)bn * T_ * DM_;
    for (int i4 = threadIdx.x * 4; i4 < T_ * DM_; i4 += 256 * 4) {
        const float4 f = *(const float4*)(sp + i4);
        xs[i4 + 0] = (__bf16)f.x; xs[i4 + 1] = (__bf16)f.y;
        xs[i4 + 2] = (__bf16)f.z; xs[i4 + 3] = (__bf16)f.w;
    }
    for (int i = threadIdx.x * 2; i < DM_; i += 256 * 2) {
        xs[T_ * DM_ + i]     = (__bf16)0.0f;
        xs[T_ * DM_ + i + 1] = (__bf16)0.0f;
    }
    __syncthreads();

    const int wave = threadIdx.x >> 5;
    const int lane = threadIdx.x & 31;
    const int colL = lane & 15;
    const int aoff = (lane < 16) ? 0 : 16;
    const int boff = (lane < 16) ? 0 : 16;
    const int hi8  = (lane < 16) ? 0 : 8;

    __bf16* dbase = dst + (size_t)bn * T_ * DM_;

    // Per (tap, row-tile): per-lane source row, redirected to zero row 48 when invalid.
    int srow[3][3];
#pragma unroll
    for (int tap = 0; tap < 3; ++tap) {
        const int shift = (ntaps == 3) ? (tap - 1) : 0;
#pragma unroll
        for (int ti = 0; ti < 3; ++ti) {
            const int trow = ti * 16 + colL;
            const int sr   = trow + shift;
            const bool valid = (sr >= 0) && (sr < T_) && (segid(sr) == segid(trow));
            srow[tap][ti] = valid ? sr : T_;   // row 48 = zeros
        }
    }

    // 32 column tiles of 16, split across 8 waves; each wave keeps 3 row-tiles live.
    for (int jj = wave; jj < 32; jj += 8) {
        const int oc = jj * 16 + colL;
        v8f acc[3];
#pragma unroll
        for (int ti = 0; ti < 3; ++ti) acc[ti] = zero8();

        for (int tap = 0; tap < ntaps; ++tap) {
            const __bf16* wrow = w + ((size_t)tap * DM_ + oc) * DM_;
            const __bf16* arow0 = xs + srow[tap][0] * DM_;
            const __bf16* arow1 = xs + srow[tap][1] * DM_;
            const __bf16* arow2 = xs + srow[tap][2] * DM_;
            for (int ki = 0; ki < 16; ++ki) {
                const int kb = ki * 32;
                v16bf bf = load_bfrag(wrow, kb, boff);
                v16bf a0 = load_afrag(arow0, kb, aoff);
                v16bf a1 = load_afrag(arow1, kb, aoff);
                v16bf a2 = load_afrag(arow2, kb, aoff);
                acc[0] = __builtin_amdgcn_wmma_f32_16x16x32_bf16(
                    false, a0, false, bf, (short)0, acc[0], false, false);
                acc[1] = __builtin_amdgcn_wmma_f32_16x16x32_bf16(
                    false, a1, false, bf, (short)0, acc[1], false, false);
                acc[2] = __builtin_amdgcn_wmma_f32_16x16x32_bf16(
                    false, a2, false, bf, (short)0, acc[2], false, false);
            }
        }

        const float bv = bias[oc];
#pragma unroll
        for (int ti = 0; ti < 3; ++ti) {
#pragma unroll
            for (int e = 0; e < 8; ++e) {
                const int t = ti * 16 + e + hi8;
                dbase[(size_t)t * DM_ + oc] = (__bf16)(acc[ti][e] + bv);
            }
        }
    }
}

// ---------------------------------------------------------------------------
// Kernel 2: fused attention (scores, mask, softmax, attn@V) + output projection.
// One workgroup per (b,n); one wave per head; then all 8 waves do the proj GEMM.
// ---------------------------------------------------------------------------
__global__ __launch_bounds__(256) void attn_proj(
    const __bf16* __restrict__ qb, const __bf16* __restrict__ kb,
    const __bf16* __restrict__ vb,
    const int* __restrict__ mask,
    const __bf16* __restrict__ wpj, const float* __restrict__ pjb,
    float* __restrict__ out)
{
    __shared__ __attribute__((aligned(16))) __bf16 Ppad[H_][T_][64];   // 48KB (K padded 48->64)
    __shared__ __attribute__((aligned(16))) __bf16 ctx[T_][DM_];       // 48KB

    const int bn = blockIdx.x;
    const int b  = bn / N_;
    const int wave = threadIdx.x >> 5;       // = head
    const int lane = threadIdx.x & 31;
    const int colL = lane & 15;
    const int aoff = (lane < 16) ? 0 : 16;
    const int boff = (lane < 16) ? 0 : 16;
    const int hi8  = (lane < 16) ? 0 : 8;
    const int h = wave;

    const __bf16* qh = qb + (size_t)bn * T_ * DM_ + h * HD_;
    const __bf16* kh = kb + (size_t)bn * T_ * DM_ + h * HD_;
    const __bf16* vh = vb + (size_t)bn * T_ * DM_ + h * HD_;

    // ---- scores: S[t][s] = sum_d q[t,d]*k[s,d] ----
    v16bf qfr[2][3];
#pragma unroll
    for (int kc = 0; kc < 2; ++kc)
#pragma unroll
        for (int ti = 0; ti < 3; ++ti)
            qfr[kc][ti] = load_afrag(qh + (size_t)(ti * 16 + colL) * DM_, kc * 32, aoff);

    v8f S[3][3];
#pragma unroll
    for (int ti = 0; ti < 3; ++ti)
#pragma unroll
        for (int tj = 0; tj < 3; ++tj) S[ti][tj] = zero8();

#pragma unroll
    for (int tj = 0; tj < 3; ++tj) {
#pragma unroll
        for (int kc = 0; kc < 2; ++kc) {
            v16bf bf = load_bfrag(kh + (size_t)(tj * 16 + colL) * DM_, kc * 32, boff);
#pragma unroll
            for (int ti = 0; ti < 3; ++ti)
                S[ti][tj] = __builtin_amdgcn_wmma_f32_16x16x32_bf16(
                    false, qfr[kc][ti], false, bf, (short)0, S[ti][tj], false, false);
        }
    }

    // ---- scale + mask ----
    const float scale = 0.125f;  // 1/sqrt(64)
#pragma unroll
    for (int ti = 0; ti < 3; ++ti)
#pragma unroll
        for (int tj = 0; tj < 3; ++tj)
#pragma unroll
            for (int e = 0; e < 8; ++e) {
                const int t = ti * 16 + e + hi8;
                const int s = tj * 16 + colL;
                float x = S[ti][tj][e] * scale;
                if (mask[((size_t)b * T_ + t) * T_ + s] == 0) x = -1.0e9f;
                S[ti][tj][e] = x;
            }

    // ---- softmax over s (rows live in 16-lane groups, 3 regs per row) ----
#pragma unroll
    for (int ti = 0; ti < 3; ++ti) {
#pragma unroll
        for (int e = 0; e < 8; ++e) {
            float m = fmaxf(S[ti][0][e], fmaxf(S[ti][1][e], S[ti][2][e]));
            m = fmaxf(m, __shfl_xor(m, 1, 32));
            m = fmaxf(m, __shfl_xor(m, 2, 32));
            m = fmaxf(m, __shfl_xor(m, 4, 32));
            m = fmaxf(m, __shfl_xor(m, 8, 32));
            float p0 = __expf(S[ti][0][e] - m);
            float p1 = __expf(S[ti][1][e] - m);
            float p2 = __expf(S[ti][2][e] - m);
            float sum = p0 + p1 + p2;
            sum += __shfl_xor(sum, 1, 32);
            sum += __shfl_xor(sum, 2, 32);
            sum += __shfl_xor(sum, 4, 32);
            sum += __shfl_xor(sum, 8, 32);
            const float inv = 1.0f / sum;
            const int t = ti * 16 + e + hi8;
            Ppad[wave][t][ 0 + colL] = (__bf16)(p0 * inv);
            Ppad[wave][t][16 + colL] = (__bf16)(p1 * inv);
            Ppad[wave][t][32 + colL] = (__bf16)(p2 * inv);
        }
    }
    // zero the K-padding region s=48..63
    for (int idx = lane; idx < T_ * 16; idx += 32)
        Ppad[wave][idx >> 4][48 + (idx & 15)] = (__bf16)0.0f;

    // ---- ctx = P @ V  (48x48pad64 @ 48x64) ----
    v16bf pfr[2][3];
#pragma unroll
    for (int kc = 0; kc < 2; ++kc)
#pragma unroll
        for (int ti = 0; ti < 3; ++ti)
            pfr[kc][ti] = load_afrag(&Ppad[wave][ti * 16 + colL][0], kc * 32, aoff);

#pragma unroll
    for (int cj = 0; cj < 4; ++cj) {
        v8f acc[3];
#pragma unroll
        for (int ti = 0; ti < 3; ++ti) acc[ti] = zero8();
#pragma unroll
        for (int kc = 0; kc < 2; ++kc) {
            v16bf bf;
#pragma unroll
            for (int e = 0; e < 16; ++e) {
                const int s = kc * 32 + boff + e;
                bf[e] = (s < T_) ? vh[(size_t)s * DM_ + cj * 16 + colL] : (__bf16)0.0f;
            }
#pragma unroll
            for (int ti = 0; ti < 3; ++ti)
                acc[ti] = __builtin_amdgcn_wmma_f32_16x16x32_bf16(
                    false, pfr[kc][ti], false, bf, (short)0, acc[ti], false, false);
        }
#pragma unroll
        for (int ti = 0; ti < 3; ++ti)
#pragma unroll
            for (int e = 0; e < 8; ++e) {
                const int t = ti * 16 + e + hi8;
                ctx[t][h * HD_ + cj * 16 + colL] = (__bf16)acc[ti][e];
            }
    }

    __syncthreads();

    // ---- output projection: out = ctx @ wproj^T + b  (48x512x512) ----
    float* obase = out + (size_t)bn * T_ * DM_;
    for (int tl = wave; tl < 96; tl += 8) {
        const int ti = tl % 3;
        const int jj = tl / 3;
        const int oc = jj * 16 + colL;
        const __bf16* wrow = wpj + (size_t)oc * DM_;
        v8f acc = zero8();
        for (int kc = 0; kc < 16; ++kc) {
            v16bf bf = load_bfrag(wrow, kc * 32, boff);
            v16bf af = load_afrag(&ctx[ti * 16 + colL][0], kc * 32, aoff);
            acc = __builtin_amdgcn_wmma_f32_16x16x32_bf16(
                false, af, false, bf, (short)0, acc, false, false);
        }
        const float bv = pjb[oc];
#pragma unroll
        for (int e = 0; e < 8; ++e) {
            const int t = ti * 16 + e + hi8;
            obase[(size_t)t * DM_ + oc] = acc[e] + bv;
        }
    }
}

// ---------------------------------------------------------------------------
extern "C" void kernel_launch(void* const* d_in, const int* in_sizes, int n_in,
                              void* d_out, int out_size, void* d_ws, size_t ws_size,
                              hipStream_t stream) {
    const float* query  = (const float*)d_in[0];
    const float* key_in = (const float*)d_in[1];
    const float* value  = (const float*)d_in[2];
    const int*   mask   = (const int*)  d_in[3];
    const float* cqw    = (const float*)d_in[4];
    const float* cqb    = (const float*)d_in[5];
    const float* ckw    = (const float*)d_in[6];
    const float* ckb    = (const float*)d_in[7];
    const float* fcw    = (const float*)d_in[8];
    const float* fcb    = (const float*)d_in[9];
    const float* pjw    = (const float*)d_in[10];
    const float* pjb    = (const float*)d_in[11];

    char* ws = (char*)d_ws;
    const size_t CONVW = (size_t)3 * DM_ * DM_ * 2;      // 1,572,864 B
    const size_t SQW   = (size_t)DM_ * DM_ * 2;          //   524,288 B
    const size_t TILE  = (size_t)B_ * N_ * T_ * DM_ * 2; // 117,964,800 B
    __bf16* wq  = (__bf16*)(ws);
    __bf16* wk  = (__bf16*)(ws + CONVW);
    __bf16* wfc = (__bf16*)(ws + 2 * CONVW);
    __bf16* wpj = (__bf16*)(ws + 2 * CONVW + SQW);
    __bf16* qbp = (__bf16*)(ws + 2 * CONVW + 2 * SQW);
    __bf16* kbp = (__bf16*)(ws + 2 * CONVW + 2 * SQW + TILE);
    __bf16* vbp = (__bf16*)(ws + 2 * CONVW + 2 * SQW + 2 * TILE);

    repack_weights<<<dim3(3072, 4), 256, 0, stream>>>(cqw, ckw, fcw, pjw, wq, wk, wfc, wpj);
    proj_qkv<<<B_ * N_ * 3, 256, 0, stream>>>(query, key_in, value, cqb, ckb, fcb,
                                              wq, wk, wfc, qbp, kbp, vbp);
    attn_proj<<<B_ * N_, 256, 0, stream>>>(qbp, kbp, vbp, mask, wpj, pjb, (float*)d_out);
}
